// AttentionLayer_21577915695116
// MI455X (gfx1250) — matrix operand order repaired and checked by reading, not compiled, not run on hardware
//
#include <hip/hip_runtime.h>
#include <hip/hip_bf16.h>
#include <cstdint>

typedef __attribute__((ext_vector_type(16))) __bf16 v16bf;
typedef __attribute__((ext_vector_type(8)))  float  v8f;

typedef __attribute__((ext_vector_type(4))) unsigned int tdm_v4u;
typedef __attribute__((ext_vector_type(8))) int          tdm_v8i;
typedef __attribute__((ext_vector_type(4))) int          tdm_v4i;

#define BDIM   4
#define SEQ    2048
#define DMODEL 1024
#define NHEAD  16
#define HDIM   64
#define MTOT   (BDIM * SEQ)   // 8192

#if defined(__AMDGCN__) && __has_builtin(__builtin_amdgcn_tensor_load_to_lds)
#define HAVE_TDM 1
#else
#define HAVE_TDM 0
#endif

__device__ __forceinline__ __bf16 f2bf(float f) {
  union { float f; unsigned u; } v; v.f = f;
  unsigned r = v.u + 0x7FFFu + ((v.u >> 16) & 1u);   // round-to-nearest-even
  unsigned short s = (unsigned short)(r >> 16);
  __bf16 b; __builtin_memcpy(&b, &s, 2); return b;
}

// A-matrix fragment, 16x32 bf16 (ISA 7.12.2): lane m=lane&15; halves 0..7 = K kh..kh+7,
// halves 8..15 = K 16+kh..23+kh, kh = (lane>=16)*8.
__device__ __forceinline__ v16bf frag_a(const __bf16* base, int ld) {
  int lane = threadIdx.x & 31;
  int r  = lane & 15;
  int kh = (lane >> 4) << 3;
  const __bf16* p = base + r * ld + kh;
  v16bf f;
#pragma unroll
  for (int i = 0; i < 8; ++i) f[i] = p[i];
#pragma unroll
  for (int i = 0; i < 8; ++i) f[8 + i] = p[16 + i];
  return f;
}

// B-matrix fragment, 32x16 bf16: lane n=lane&15; halves 0..7 = K kh..kh+7,
// halves 8..15 = K 16+kh..23+kh, kh = (lane>=16)*8. Stored as [n][k] rows in LDS.
__device__ __forceinline__ v16bf frag_b(const __bf16* base, int ld) {
  int lane = threadIdx.x & 31;
  int n  = lane & 15;
  int k0 = (lane >> 4) << 3;
  const __bf16* p = base + n * ld + k0;
  v16bf f;
#pragma unroll
  for (int i = 0; i < 8; ++i) f[i] = p[i];
#pragma unroll
  for (int i = 0; i < 8; ++i) f[8 + i] = p[16 + i];
  return f;
}

#define WMMA_BF16(a, b, c) \
  __builtin_amdgcn_wmma_f32_16x16x32_bf16(false, (a), false, (b), (short)0, (c), false, false)

// ---------------------------------------------------------------------------
// TDM: load a 64x64 bf16 tile (global row stride DMODEL elements) into LDS
// with 8-element (4-DWORD) padding after each 64-element (32-DWORD) row,
// matching an LDS leading dimension of 72 bf16.
// D# per ISA ch.8: group0 {count, lds_addr, global_addr, type=2},
// group1 {data_size=2B, pad_enable, pad_interval=32dw, pad_amount=4dw,
//         tensor_dim0/1=64, tile_dim0/1=64, tensor_dim0_stride=DMODEL}.
// Issue from ONE wave only; completion via s_wait_tensorcnt.
// ---------------------------------------------------------------------------
__device__ __forceinline__ void tdm_load_tile64(const void* gptr, void* lptr) {
#if HAVE_TDM
  uint64_t ga = (uint64_t)(uintptr_t)gptr;
  uint32_t la = (uint32_t)(uintptr_t)lptr;
  tdm_v4u g0;
  g0[0] = 1u;                                              // count=1, user D#
  g0[1] = la;                                              // lds_addr (bytes)
  g0[2] = (uint32_t)ga;                                    // global_addr[31:0]
  g0[3] = (uint32_t)((ga >> 32) & 0x01FFFFFFu) | (2u << 30); // ga[56:32], type=2
  tdm_v8i g1;
  g1[0] = (1 << 16)    // data_size = 2 bytes
        | (1 << 20)    // pad_enable
        | (4 << 22)    // pad_interval: 2^(4+1) = 32 DWORDs (= 64 bf16 row)
        | (3 << 25);   // pad_amount: 3+1 = 4 DWORDs (= 8 bf16)
  g1[1] = (int)(64u << 16);   // tensor_dim0[15:0] = 64 (in [31:16])
  g1[2] = (int)(64u << 16);   // tensor_dim0[31:16]=0 | tensor_dim1[15:0]=64
  g1[3] = (int)(64u << 16);   // tensor_dim1[31:16]=0 | tile_dim0=64
  g1[4] = 64;                 // tile_dim1=64, tile_dim2=0
  g1[5] = DMODEL;             // tensor_dim0_stride[31:0] (data_size units)
  g1[6] = 0;                  // stride0 hi | stride1 lo
  g1[7] = 0;                  // stride1 hi
  tdm_v4i g2 = {};            // 2-D tensor: groups 2/3 unused
  tdm_v4i g3 = {};
#if __clang_major__ >= 23
  tdm_v8i g4 = {};
  __builtin_amdgcn_tensor_load_to_lds(g0, g1, g2, g3, g4, 0);
#else
  __builtin_amdgcn_tensor_load_to_lds(g0, g1, g2, g3, 0);
#endif
#endif
}

// ---------------------------------------------------------------------------
// Kernel 1: fused QKV projection  out = X @ W^T + bias  (NT GEMM), bf16 out
// Block tile 128x64, K staged 64 at a time. 8 waves, each 32x32.
// ---------------------------------------------------------------------------
#define K1_TM 128
#define K1_TN 64
#define K1_TK 64
#define K1_LD (K1_TK + 8)

__global__ __launch_bounds__(256)
void qkv_proj_kernel(const float* __restrict__ q_in, const float* __restrict__ k_in,
                     const float* __restrict__ v_in,
                     const float* __restrict__ Wq, const float* __restrict__ bq,
                     const float* __restrict__ Wk, const float* __restrict__ bk,
                     const float* __restrict__ Wv, const float* __restrict__ bv,
                     __bf16* __restrict__ q_o, __bf16* __restrict__ k_o,
                     __bf16* __restrict__ v_o) {
  __shared__ __align__(16) __bf16 At[K1_TM * K1_LD];
  __shared__ __align__(16) __bf16 Bt[K1_TN * K1_LD];

  const float* X; const float* W; const float* bias; __bf16* out;
  if (blockIdx.z == 0)      { X = q_in; W = Wq; bias = bq; out = q_o; }
  else if (blockIdx.z == 1) { X = k_in; W = Wk; bias = bk; out = k_o; }
  else                      { X = v_in; W = Wv; bias = bv; out = v_o; }

  const int m0 = blockIdx.x * K1_TM;
  const int n0 = blockIdx.y * K1_TN;
  const int tid  = threadIdx.x;
  const int wave = tid >> 5;
  const int lane = tid & 31;
  const int m0w = (wave >> 1) * 32;
  const int n0w = (wave & 1) * 32;

  v8f acc[2][2] = {};

  for (int k0 = 0; k0 < DMODEL; k0 += K1_TK) {
    // Prefetch next K-slab into caches while this one computes.
    if (k0 + K1_TK < DMODEL) {
      __builtin_prefetch(X + (size_t)(m0 + (tid >> 1)) * DMODEL + (k0 + K1_TK) + (tid & 1) * 32, 0, 1);
      if (tid < 128)
        __builtin_prefetch(W + (size_t)(n0 + (tid >> 1)) * DMODEL + (k0 + K1_TK) + (tid & 1) * 32, 0, 1);
    }
    // Stage A: 128x64 f32 -> bf16 (2048 float4, 8/thread)
#pragma unroll
    for (int i = 0; i < 8; ++i) {
      int lin = tid + 256 * i;
      int row = lin >> 4;
      int c4  = (lin & 15) * 4;
      float4 v = *(const float4*)(X + (size_t)(m0 + row) * DMODEL + k0 + c4);
      __bf16* d = &At[row * K1_LD + c4];
      d[0] = f2bf(v.x); d[1] = f2bf(v.y); d[2] = f2bf(v.z); d[3] = f2bf(v.w);
    }
    // Stage B: 64x64 f32 -> bf16 (1024 float4, 4/thread)
#pragma unroll
    for (int i = 0; i < 4; ++i) {
      int lin = tid + 256 * i;
      int row = lin >> 4;
      int c4  = (lin & 15) * 4;
      float4 v = *(const float4*)(W + (size_t)(n0 + row) * DMODEL + k0 + c4);
      __bf16* d = &Bt[row * K1_LD + c4];
      d[0] = f2bf(v.x); d[1] = f2bf(v.y); d[2] = f2bf(v.z); d[3] = f2bf(v.w);
    }
    __syncthreads();
#pragma unroll
    for (int ks = 0; ks < K1_TK; ks += 32) {
      v16bf a0 = frag_a(&At[(m0w +  0) * K1_LD + ks], K1_LD);
      v16bf a1 = frag_a(&At[(m0w + 16) * K1_LD + ks], K1_LD);
      v16bf b0 = frag_b(&Bt[(n0w +  0) * K1_LD + ks], K1_LD);
      v16bf b1 = frag_b(&Bt[(n0w + 16) * K1_LD + ks], K1_LD);
      acc[0][0] = WMMA_BF16(a0, b0, acc[0][0]);
      acc[0][1] = WMMA_BF16(a0, b1, acc[0][1]);
      acc[1][0] = WMMA_BF16(a1, b0, acc[1][0]);
      acc[1][1] = WMMA_BF16(a1, b1, acc[1][1]);
    }
    __syncthreads();
  }

  const int hi = lane >> 4, lo = lane & 15;
#pragma unroll
  for (int mt = 0; mt < 2; ++mt)
#pragma unroll
    for (int nt = 0; nt < 2; ++nt) {
      int col = n0 + n0w + nt * 16 + lo;
      float bb = bias[col];
#pragma unroll
      for (int r = 0; r < 8; ++r) {
        int row = m0 + m0w + mt * 16 + r + hi * 8;
        out[(size_t)row * DMODEL + col] = f2bf(acc[mt][nt][r] + bb);
      }
    }
}

// ---------------------------------------------------------------------------
// Kernel 2: flash attention, one workgroup per (b, h, 64 query rows).
// 8 waves: wave grid 4x2 over the 64x64 S/O tiles (wave = 16 rows x 32 cols).
// Q/K tiles staged by the Tensor Data Mover (wave 0); V transposed manually.
// ---------------------------------------------------------------------------
#define A_LD 72

__global__ __launch_bounds__(256)
void attention_kernel(const __bf16* __restrict__ Q, const __bf16* __restrict__ K,
                      const __bf16* __restrict__ V, float* __restrict__ out) {
  __shared__ __align__(16) __bf16 Qt[64 * A_LD];
  __shared__ __align__(16) __bf16 Kt[64 * A_LD];
  __shared__ __align__(16) __bf16 Vt[64 * A_LD];   // transposed: [d][key]
  __shared__ __align__(16) __bf16 Pb[64 * A_LD];
  __shared__ float Sf[64 * 64];
  __shared__ float mrow[64], lrow[64], crow[64];

  const int s0 = blockIdx.x * 64;
  const int h  = blockIdx.y;
  const int b  = blockIdx.z;
  const int tid  = threadIdx.x;
  const int wave = tid >> 5;
  const int lane = tid & 31;
  const int hi = lane >> 4, lo = lane & 15;
  const int wm = (wave >> 1) * 16;
  const int wn = (wave & 1) * 32;

  const size_t rowbase = (size_t)(b * SEQ) * DMODEL + h * HDIM;

#if HAVE_TDM
  if (wave == 0) tdm_load_tile64(Q + rowbase + (size_t)s0 * DMODEL, &Qt[0]);
#else
#pragma unroll
  for (int i = 0; i < 2; ++i) {
    int lin = tid + 256 * i;
    int row = lin >> 3;
    int c   = (lin & 7) * 8;
    *(uint4*)&Qt[row * A_LD + c] =
        *(const uint4*)(Q + rowbase + (size_t)(s0 + row) * DMODEL + c);
  }
#endif
  if (tid < 64) { mrow[tid] = -3.0e38f; lrow[tid] = 0.0f; }

  v8f oacc[2] = {};

  for (int kb = 0; kb < SEQ; kb += 64) {
    __syncthreads();  // Qt/init visible (1st iter); Kt/Vt/Pb free for restage
#if HAVE_TDM
    if (wave == 0) tdm_load_tile64(K + rowbase + (size_t)kb * DMODEL, &Kt[0]);
#else
#pragma unroll
    for (int i = 0; i < 2; ++i) {
      int lin = tid + 256 * i;
      int row = lin >> 3;
      int c   = (lin & 7) * 8;
      *(uint4*)&Kt[row * A_LD + c] =
          *(const uint4*)(K + rowbase + (size_t)(kb + row) * DMODEL + c);
    }
#endif
    // Stage V tile, transposed -> Vt[d][key]
#pragma unroll
    for (int i = 0; i < 16; ++i) {
      int lin = tid + 256 * i;
      int key = lin >> 6;
      int d   = lin & 63;
      Vt[d * A_LD + key] = V[rowbase + (size_t)(kb + key) * DMODEL + d];
    }
#if HAVE_TDM
    if (wave == 0) __builtin_amdgcn_s_wait_tensorcnt(0);  // Q (1st iter) + K done
#endif
    __syncthreads();

    // S = (Q . K^T) * 1/sqrt(HD)
    v8f sacc[2] = {};
#pragma unroll
    for (int ks = 0; ks < HDIM; ks += 32) {
      v16bf qa  = frag_a(&Qt[wm * A_LD + ks], A_LD);
      v16bf kf0 = frag_b(&Kt[(wn +  0) * A_LD + ks], A_LD);
      v16bf kf1 = frag_b(&Kt[(wn + 16) * A_LD + ks], A_LD);
      sacc[0] = WMMA_BF16(qa, kf0, sacc[0]);
      sacc[1] = WMMA_BF16(qa, kf1, sacc[1]);
    }
#pragma unroll
    for (int nt = 0; nt < 2; ++nt)
#pragma unroll
      for (int r = 0; r < 8; ++r)
        Sf[(wm + r + hi * 8) * 64 + wn + nt * 16 + lo] = sacc[nt][r] * 0.125f;
    __syncthreads();

    // Online softmax: 4 threads per row, wave32 shfl reductions
    {
      int row  = tid >> 2;
      int part = (tid & 3) * 16;
      float mold = mrow[row];
      float lm = -3.0e38f;
#pragma unroll
      for (int c = 0; c < 16; ++c) lm = fmaxf(lm, Sf[row * 64 + part + c]);
      lm = fmaxf(lm, __shfl_xor(lm, 1, 32));
      lm = fmaxf(lm, __shfl_xor(lm, 2, 32));
      float mnew = fmaxf(mold, lm);
      float psum = 0.0f;
#pragma unroll
      for (int c = 0; c < 16; ++c) {
        float e = __expf(Sf[row * 64 + part + c] - mnew);
        psum += e;
        Pb[row * A_LD + part + c] = f2bf(e);
      }
      psum += __shfl_xor(psum, 1, 32);
      psum += __shfl_xor(psum, 2, 32);
      if ((tid & 3) == 0) {
        float corr = __expf(mold - mnew);
        lrow[row] = lrow[row] * corr + psum;
        mrow[row] = mnew;
        crow[row] = corr;
      }
    }
    __syncthreads();

    // Rescale O by per-row correction, then O += P . V
#pragma unroll
    for (int nt = 0; nt < 2; ++nt)
#pragma unroll
      for (int r = 0; r < 8; ++r)
        oacc[nt][r] *= crow[wm + r + hi * 8];
#pragma unroll
    for (int ks = 0; ks < HDIM; ks += 32) {
      v16bf pa  = frag_a(&Pb[wm * A_LD + ks], A_LD);
      v16bf vf0 = frag_b(&Vt[(wn +  0) * A_LD + ks], A_LD);
      v16bf vf1 = frag_b(&Vt[(wn + 16) * A_LD + ks], A_LD);
      oacc[0] = WMMA_BF16(pa, vf0, oacc[0]);
      oacc[1] = WMMA_BF16(pa, vf1, oacc[1]);
    }
  }
  __syncthreads();

#pragma unroll
  for (int nt = 0; nt < 2; ++nt) {
    int col = h * HDIM + wn + nt * 16 + lo;
#pragma unroll
    for (int r = 0; r < 8; ++r) {
      int row = wm + r + hi * 8;
      out[(size_t)(b * SEQ + s0 + row) * DMODEL + col] = oacc[nt][r] / lrow[row];
    }
  }
}

// ---------------------------------------------------------------------------
extern "C" void kernel_launch(void* const* d_in, const int* in_sizes, int n_in,
                              void* d_out, int out_size, void* d_ws, size_t ws_size,
                              hipStream_t stream) {
  const float* q_in = (const float*)d_in[0];
  const float* k_in = (const float*)d_in[1];
  const float* v_in = (const float*)d_in[2];
  const float* Wq   = (const float*)d_in[3];
  const float* bq   = (const float*)d_in[4];
  const float* Wk   = (const float*)d_in[5];
  const float* bk   = (const float*)d_in[6];
  const float* Wv   = (const float*)d_in[7];
  const float* bv   = (const float*)d_in[8];

  __bf16* q_ws = (__bf16*)d_ws;
  __bf16* k_ws = q_ws + (size_t)MTOT * DMODEL;
  __bf16* v_ws = k_ws + (size_t)MTOT * DMODEL;

  dim3 g1(MTOT / K1_TM, DMODEL / K1_TN, 3);
  qkv_proj_kernel<<<g1, 256, 0, stream>>>(q_in, k_in, v_in, Wq, bq, Wk, bk, Wv, bv,
                                          q_ws, k_ws, v_ws);

  dim3 g2(SEQ / 64, NHEAD, BDIM);
  attention_kernel<<<g2, 256, 0, stream>>>(q_ws, k_ws, v_ws, (float*)d_out);
}
